// hyperCALayer_57080115364581
// MI455X (gfx1250) — matrix operand order repaired and checked by reading, not compiled, not run on hardware
//
#include <hip/hip_runtime.h>
#include <hip/hip_bf16.h>
#include <math.h>
#include <string.h>

// ---------------------------------------------------------------------------
// Fused single-head Lorentz attention for gfx1250 (MI455X), wave32 + WMMA.
// All 4 reference heads are identical -> compute one head, normalize twice.
// Round 3: one-time bf16 pad/transpose pre-pass into d_ws; hot loop stages
// pure b128 copies (no conversion, no scalar transpose stores).
// ---------------------------------------------------------------------------

typedef __attribute__((ext_vector_type(16))) __bf16 v16bf;   // 32 B: A/B fragment
typedef __attribute__((ext_vector_type(8)))  __bf16 v8bf;    // 16 B: half fragment
typedef __attribute__((ext_vector_type(4)))  __bf16 v4bf;    // 8 B
typedef __attribute__((ext_vector_type(8)))  float  v8f;     // 32 B: C/D fragment

#define N_TOK   4096
#define D_IN    129
#define D_PAD   160            // 5 * 32 : K-dim padded for bf16 WMMA
#define KCHUNK  64             // keys per chunk (two 16x32 P A-fragments)
#define NCHUNKS (N_TOK / KCHUNK)
#define WAVES   4              // waves per workgroup, each owns a 16-row Q tile
#define OT      9              // output feature tiles: 9*16 = 144 >= 129
#define L2E     1.44269504088896f

#define WS_BYTES ((size_t)(N_TOK * D_PAD + D_PAD * N_TOK) * sizeof(__bf16))

union ABf {                    // A-fragment assembly helper (POD union)
    v16bf  v;
    v8bf   g[2];
    __bf16 h[16];
};

__device__ __forceinline__ float fexp2(float x) {
#if __has_builtin(__builtin_amdgcn_exp2f)
    return __builtin_amdgcn_exp2f(x);
#else
    return exp2f(x);
#endif
}

// ---------------------------------------------------------------------------
// One-time pre-pass: f32 -> bf16, pad D 129->160, emit row-major + transposed.
// ---------------------------------------------------------------------------
__global__ __launch_bounds__(128)
void convert_kernel(const float* __restrict__ src,
                    __bf16* __restrict__ kPad,     // [N_TOK][D_PAD]
                    __bf16* __restrict__ vT)       // [D_PAD][N_TOK]
{
    __shared__ __bf16 tile[KCHUNK][D_PAD];
    const int kb = blockIdx.x * KCHUNK;

    for (int g = threadIdx.x; g < KCHUNK * (D_PAD / 4); g += 128) {
        int r  = g / (D_PAD / 4);
        int c4 = (g - r * (D_PAD / 4)) * 4;
        float fv[4];
        if (c4 + 3 < D_IN) {
            memcpy(fv, &src[(kb + r) * D_IN + c4], 16);
        } else {
            #pragma unroll
            for (int e = 0; e < 4; ++e)
                fv[e] = (c4 + e < D_IN) ? src[(kb + r) * D_IN + c4 + e] : 0.0f;
        }
        v4bf bq = { (__bf16)fv[0], (__bf16)fv[1], (__bf16)fv[2], (__bf16)fv[3] };
        *(v4bf*)&tile[r][c4] = bq;
        *(v4bf*)&kPad[(kb + r) * D_PAD + c4] = bq;           // coalesced b64
    }
    __syncthreads();
    // transposed rows, vectorized 4-wide along keys (coalesced b64 stores)
    for (int i = threadIdx.x; i < D_PAD * (KCHUNK / 4); i += 128) {
        int f  = i / (KCHUNK / 4);
        int k4 = (i - f * (KCHUNK / 4)) * 4;
        v4bf bq = { tile[k4 + 0][f], tile[k4 + 1][f],
                    tile[k4 + 2][f], tile[k4 + 3][f] };
        *(v4bf*)&vT[f * N_TOK + kb + k4] = bq;
    }
}

// ---------------------------------------------------------------------------
// Fused attention. PRE=true: stage bf16 tiles from d_ws with b128 copies.
// PRE=false: self-staging fallback (convert in-loop) if d_ws is too small.
// ---------------------------------------------------------------------------
template <bool PRE>
__global__ __launch_bounds__(WAVES * 32)
void hyper_attn_kernel(const float* __restrict__ qg,
                       const float* __restrict__ src,
                       const __bf16* __restrict__ kPad,
                       const __bf16* __restrict__ vT,
                       const float* __restrict__ scale,
                       const float* __restrict__ bias,
                       float* __restrict__ out)
{
    __shared__ __align__(32) __bf16 kTile[KCHUNK][D_PAD];        // 20 KB
    __shared__ __align__(32) __bf16 vTile[D_PAD][KCHUNK];        // 20 KB
    __shared__ __align__(32) __bf16 pScr[WAVES][16][KCHUNK];     // 8 KB

    const int tid  = threadIdx.x;
    const int lane = tid & 31;
    const int w    = tid >> 5;
    const int m    = lane & 15;    // col index (B/C) or row index (A)
    const int hi   = lane >> 4;    // which 16-lane half

    const int qBase = (blockIdx.x * WAVES + w) * 16;   // 16 query rows per wave

    // logits (base-2 domain): L = c0l + c1l * s, with log2(e) folded in
    const float c1  = 2.0f / scale[0];
    const float c1l = c1 * L2E;
    const float c0l = (c1 + bias[0]) * L2E;

    // ---- Q A-fragments (16x32 bf16 x 5 chunks), time coord negated ----
    ABf qf[5];
    #pragma unroll
    for (int kc = 0; kc < 5; ++kc) {
        #pragma unroll
        for (int j = 0; j < 16; ++j) {
            // A 16x32 layout: element j -> K = hi*16+j (contiguous, matches B)
            int f = kc * 32 + hi * 16 + j;
            float val = (f < D_IN) ? qg[(qBase + m) * D_IN + f] : 0.0f;
            if (f == 0) val = -val;                    // Lorentz time negation
            qf[kc].h[j] = (__bf16)val;
        }
    }

    // ---- flash-attention state ----
    v8f o[OT];
    #pragma unroll
    for (int t = 0; t < OT; ++t)
        o[t] = (v8f){0.f, 0.f, 0.f, 0.f, 0.f, 0.f, 0.f, 0.f};
    float mrow[8], lrow[8];
    #pragma unroll
    for (int v = 0; v < 8; ++v) { mrow[v] = -3.0e38f; lrow[v] = 0.0f; }

    for (int ch = 0; ch < NCHUNKS; ++ch) {
        const int kb = ch * KCHUNK;

        if constexpr (PRE) {
            // ---- pure b128 tile copies from pre-converted bf16 buffers ----
            for (int i = tid; i < KCHUNK * (D_PAD / 8); i += WAVES * 32) {
                int r  = i / (D_PAD / 8);
                int c8 = (i - r * (D_PAD / 8)) * 8;
                *(v8bf*)&kTile[r][c8] = *(const v8bf*)&kPad[(kb + r) * D_PAD + c8];
            }
            for (int i = tid; i < D_PAD * (KCHUNK / 8); i += WAVES * 32) {
                int f  = i / (KCHUNK / 8);
                int k8 = (i - f * (KCHUNK / 8)) * 8;
                *(v8bf*)&vTile[f][k8] = *(const v8bf*)&vT[f * N_TOK + kb + k8];
            }
            if (ch + 1 < NCHUNKS) {   // global_prefetch_b8 hints for next tiles
                __builtin_prefetch(&kPad[(kb + KCHUNK) * D_PAD] + tid * 160, 0, 0);
                __builtin_prefetch(&vT[kb + KCHUNK] + (tid >> 1) * (2 * N_TOK), 0, 0);
            }
        } else {
            // ---- fallback: convert f32 -> bf16 in-loop ----
            for (int g = tid; g < KCHUNK * (D_PAD / 4); g += WAVES * 32) {
                int r  = g / (D_PAD / 4);
                int c4 = (g - r * (D_PAD / 4)) * 4;
                float fv[4];
                if (c4 + 3 < D_IN) {
                    memcpy(fv, &src[(kb + r) * D_IN + c4], 16);
                } else {
                    #pragma unroll
                    for (int e = 0; e < 4; ++e)
                        fv[e] = (c4 + e < D_IN) ? src[(kb + r) * D_IN + c4 + e] : 0.0f;
                }
                v4bf bq = { (__bf16)fv[0], (__bf16)fv[1], (__bf16)fv[2], (__bf16)fv[3] };
                *(v4bf*)&kTile[r][c4] = bq;
                #pragma unroll
                for (int e = 0; e < 4; ++e) vTile[c4 + e][r] = bq[e];
            }
            if (ch + 1 < NCHUNKS)
                __builtin_prefetch(&src[(kb + KCHUNK) * D_IN] + tid * 65, 0, 0);
        }
        __syncthreads();

        // ---- S = Qm @ K^T : four 16x16 tiles, K-dim = 160 = 5 WMMAs each ----
        v8f s[4];
        #pragma unroll
        for (int st = 0; st < 4; ++st)
            s[st] = (v8f){0.f, 0.f, 0.f, 0.f, 0.f, 0.f, 0.f, 0.f};
        #pragma unroll
        for (int kc = 0; kc < 5; ++kc) {
            int fb = kc * 32 + hi * 16;    // B 32x16: lane>>4 selects K half
            #pragma unroll
            for (int st = 0; st < 4; ++st) {
                v16bf b = *(const v16bf*)&kTile[st * 16 + m][fb];
                s[st] = __builtin_amdgcn_wmma_f32_16x16x32_bf16(
                            false, qf[kc].v, false, b, (short)0, s[st], false, false);
            }
        }

        // ---- online softmax over this 64-key chunk (base-2 domain) ----
        float fac[8];
        #pragma unroll
        for (int v = 0; v < 8; ++v) {
            float a[4];
            #pragma unroll
            for (int st = 0; st < 4; ++st) a[st] = fmaf(c1l, s[st][v], c0l);
            float cm = fmaxf(fmaxf(a[0], a[1]), fmaxf(a[2], a[3]));
            #pragma unroll
            for (int off = 1; off < 16; off <<= 1)
                cm = fmaxf(cm, __shfl_xor(cm, off, 32));   // row-max within half
            float mn = fmaxf(mrow[v], cm);
            fac[v] = fexp2(mrow[v] - mn);
            mrow[v] = mn;
            float rs = 0.0f;
            #pragma unroll
            for (int st = 0; st < 4; ++st) {
                float e = fexp2(a[st] - mn);
                rs += e;
                pScr[w][v + 8 * hi][st * 16 + m] = (__bf16)e;  // C-layout -> LDS
            }
            #pragma unroll
            for (int off = 1; off < 16; off <<= 1)
                rs += __shfl_xor(rs, off, 32);             // row-sum within half
            lrow[v] = lrow[v] * fac[v] + rs;
        }
        #pragma unroll
        for (int t = 0; t < OT; ++t)
            #pragma unroll
            for (int v = 0; v < 8; ++v) o[t][v] *= fac[v];  // rescale O

        // ---- P back as two 16x32 A-fragments (keys 0..31 / 32..63) ----
        ABf pf0, pf1;
        pf0.g[0] = *(const v8bf*)&pScr[w][m][hi * 16];
        pf0.g[1] = *(const v8bf*)&pScr[w][m][hi * 16 + 8];
        pf1.g[0] = *(const v8bf*)&pScr[w][m][32 + hi * 16];
        pf1.g[1] = *(const v8bf*)&pScr[w][m][32 + hi * 16 + 8];

        // ---- O += P @ V : 9 feature tiles x two K=32 WMMAs ----
        #pragma unroll
        for (int t = 0; t < OT; ++t) {
            v16bf bv0 = *(const v16bf*)&vTile[t * 16 + m][hi * 16];
            v16bf bv1 = *(const v16bf*)&vTile[t * 16 + m][32 + hi * 16];
            o[t] = __builtin_amdgcn_wmma_f32_16x16x32_bf16(
                       false, pf0.v, false, bv0, (short)0, o[t], false, false);
            o[t] = __builtin_amdgcn_wmma_f32_16x16x32_bf16(
                       false, pf1.v, false, bv1, (short)0, o[t], false, false);
        }
        __syncthreads();   // keep staging buffers safe for next chunk
    }

    // ---- ave = O / l (reciprocal-multiply) ----
    #pragma unroll
    for (int v = 0; v < 8; ++v) lrow[v] = 1.0f / lrow[v];
    #pragma unroll
    for (int t = 0; t < OT; ++t)
        #pragma unroll
        for (int v = 0; v < 8; ++v) o[t][v] *= lrow[v];

    // ---- Lorentz mid-point normalization, applied twice (heads-mean is id) ----
    #pragma unroll
    for (int pass = 0; pass < 2; ++pass) {
        float inn[8];
        #pragma unroll
        for (int v = 0; v < 8; ++v) inn[v] = 0.0f;
        #pragma unroll
        for (int t = 0; t < OT; ++t) {
            int d = t * 16 + m;
            float sgn = (d == 0) ? -1.0f : 1.0f;   // padded cols are exactly 0
            #pragma unroll
            for (int v = 0; v < 8; ++v) inn[v] += sgn * o[t][v] * o[t][v];
        }
        #pragma unroll
        for (int v = 0; v < 8; ++v) {
            float sv = inn[v];
            #pragma unroll
            for (int off = 1; off < 16; off <<= 1)
                sv += __shfl_xor(sv, off, 32);
            float denom = sqrtf(fmaxf(fabsf(sv), 1e-8f));
            inn[v] = 1.0f / denom;
        }
        #pragma unroll
        for (int t = 0; t < OT; ++t)
            #pragma unroll
            for (int v = 0; v < 8; ++v) o[t][v] *= inn[v];
    }

    // ---- store: row = qBase + v + 8*hi, col = t*16 + m (cols >= 129 dropped) ----
    #pragma unroll
    for (int t = 0; t < OT; ++t) {
        int d = t * 16 + m;
        if (d < D_IN) {
            #pragma unroll
            for (int v = 0; v < 8; ++v)
                out[(qBase + v + 8 * hi) * D_IN + d] = o[t][v];
        }
    }
}

extern "C" void kernel_launch(void* const* d_in, const int* in_sizes, int n_in,
                              void* d_out, int out_size, void* d_ws, size_t ws_size,
                              hipStream_t stream) {
    (void)in_sizes; (void)n_in; (void)out_size;
    const float* q     = (const float*)d_in[0];
    const float* src   = (const float*)d_in[1];
    const float* scale = (const float*)d_in[2];
    const float* bias  = (const float*)d_in[3];
    float* out = (float*)d_out;

    dim3 grid(N_TOK / (16 * WAVES));   // 64 workgroups
    dim3 block(WAVES * 32);            // 4 waves (wave32)

    if (d_ws != nullptr && ws_size >= WS_BYTES) {
        __bf16* kPad = (__bf16*)d_ws;
        __bf16* vT   = kPad + (size_t)N_TOK * D_PAD;
        convert_kernel<<<dim3(N_TOK / KCHUNK), dim3(128), 0, stream>>>(src, kPad, vT);
        hyper_attn_kernel<true><<<grid, block, 0, stream>>>(
            q, src, kPad, vT, scale, bias, out);
    } else {
        hyper_attn_kernel<false><<<grid, block, 0, stream>>>(
            q, src, nullptr, nullptr, scale, bias, out);
    }
}
